// HMultiheadAttention_45775761441162
// MI455X (gfx1250) — matrix operand order repaired and checked by reading, not compiled, not run on hardware
//
#include <hip/hip_runtime.h>
#include <math.h>

typedef _Float16 f16;
typedef _Float16 v16h __attribute__((ext_vector_type(16)));
typedef float    v8f  __attribute__((ext_vector_type(8)));

#define B_   4
#define S_   1024
#define E_   512
#define H_   8
#define HD_  64
#define EPSF 1e-7f

// ---- 16-lane (half-wave) reductions: masks 1/2/4/8 never cross lane 16, so
// lanes 0-15 and 16-31 reduce independently == per-row reduction for the
// CDNA5 WMMA C/D layout (lane n<16 holds row j, lane n>=16 holds row j+8).
__device__ __forceinline__ float hred_add16(float v) {
  v += __shfl_xor(v, 1, 32);
  v += __shfl_xor(v, 2, 32);
  v += __shfl_xor(v, 4, 32);
  v += __shfl_xor(v, 8, 32);
  return v;
}
__device__ __forceinline__ float hred_max16(float v) {
  v = fmaxf(v, __shfl_xor(v, 1, 32));
  v = fmaxf(v, __shfl_xor(v, 2, 32));
  v = fmaxf(v, __shfl_xor(v, 4, 32));
  v = fmaxf(v, __shfl_xor(v, 8, 32));
  return v;
}

// ---------------------------------------------------------------------------
// Prep 1: Zt[mat][m][k] = (f16) z_mat[k][m]   (B-fragments become contiguous)
// ---------------------------------------------------------------------------
__global__ __launch_bounds__(256) void prep_transpose_kernel(
    const float* __restrict__ zq, const float* __restrict__ zk,
    const float* __restrict__ zv, f16* __restrict__ Zt) {
  size_t idx = (size_t)blockIdx.x * 256 + threadIdx.x;
  if (idx >= (size_t)3 * E_ * E_) return;
  int mat = (int)(idx >> 18);
  int rem = (int)(idx & (E_ * E_ - 1));
  int k = rem >> 9, m = rem & (E_ - 1);
  const float* z = (mat == 0) ? zq : (mat == 1) ? zk : zv;
  Zt[((size_t)mat << 18) + ((size_t)m << 9) + k] = (f16)z[((size_t)k << 9) + m];
}

// ---------------------------------------------------------------------------
// Prep 2: zn[mat][m] = max(||z[:,m]||, eps);  ch/sh[mat][m] = cosh/sinh(2 r[m])
// ---------------------------------------------------------------------------
__global__ __launch_bounds__(256) void prep_params_kernel(
    const float* __restrict__ zq, const float* __restrict__ zk,
    const float* __restrict__ zv, const float* __restrict__ rq,
    const float* __restrict__ rk, const float* __restrict__ rv,
    float* __restrict__ zn, float* __restrict__ ch, float* __restrict__ sh) {
  int idx = blockIdx.x * 256 + threadIdx.x;
  if (idx >= 3 * E_) return;
  int mat = idx >> 9, m = idx & (E_ - 1);
  const float* z = (mat == 0) ? zq : (mat == 1) ? zk : zv;
  const float* r = (mat == 0) ? rq : (mat == 1) ? rk : rv;
  float s = 0.f;
  for (int k = 0; k < E_; ++k) { float t = z[(size_t)k * E_ + m]; s += t * t; }
  zn[idx] = fmaxf(sqrtf(s), EPSF);
  float tc = 2.f * r[m];
  ch[idx] = coshf(tc);
  sh[idx] = sinhf(tc);
}

// ---------------------------------------------------------------------------
// Prep 3: zero extended-Vt workspace (rows 65..79 must stay zero)
// ---------------------------------------------------------------------------
__global__ __launch_bounds__(256) void prep_zero_kernel(f16* __restrict__ p, size_t n) {
  size_t idx = (size_t)blockIdx.x * 256 + threadIdx.x;
  if (idx < n) p[idx] = (f16)0.f;
}

// ---------------------------------------------------------------------------
// Fused hyperbolic linear: out = h_linear(x, z, r), plus attention-side prep.
//   grid = (256 row-tiles, 3 matrices), block = 256 threads = 8 waves.
//   Wave w owns columns [64w, 64w+64) == exactly head w, so per-head sums
//   (q2/k2/lam for v) come from wave-local half-wave reductions.
// Outputs: mat0 -> pq fp32 [B,H,S,64] + q2;  mat1 -> pk + k2;
//          mat2 -> Vt_ext f16 [B,H,80,S]: rows 0..63 = lam*v (transposed),
//                  row 64 = lam-1, rows 65..79 untouched (zeroed in prep).
// ---------------------------------------------------------------------------
__global__ __launch_bounds__(256) void hlinear_kernel(
    const float* __restrict__ xq, const float* __restrict__ xk,
    const float* __restrict__ xv, const f16* __restrict__ Zt,
    const float* __restrict__ zn, const float* __restrict__ ch,
    const float* __restrict__ sh, float* __restrict__ pq,
    float* __restrict__ pk, float* __restrict__ q2a, float* __restrict__ k2a,
    f16* __restrict__ Vt) {
  int mat = blockIdx.y;
  const float* x = (mat == 0) ? xq : (mat == 1) ? xk : xv;
  const f16* Ztm = Zt + ((size_t)mat << 18);
  const float* znm = zn + mat * E_;
  const float* chm = ch + mat * E_;
  const float* shm = sh + mat * E_;

  int row0 = blockIdx.x * 16;
  int tid = threadIdx.x;

  __shared__ float red[256];
  __shared__ float lamS[16];
  __shared__ float v2S[16];

  // phase 1: x2 / lam per row
  {
    int rr = tid & 15, seg = tid >> 4;
    const float* xr = x + (size_t)(row0 + rr) * E_ + seg * 32;
    float p = 0.f;
#pragma unroll
    for (int i = 0; i < 32; ++i) { float t = xr[i]; p += t * t; }
    red[tid] = p;
  }
  __syncthreads();
  if (tid < 16) {
    float s = 0.f;
#pragma unroll
    for (int j = 0; j < 16; ++j) s += red[tid + j * 16];
    lamS[tid] = 2.f / fmaxf(1.f - s, EPSF);
    v2S[tid] = 0.f;
  }
  __syncthreads();

  int wave = tid >> 5, lane = tid & 31;
  int lrow = lane & 15, hi = lane >> 4;  // hi selects K/M half per ISA layout
  int nBase = wave * 64;

  v8f acc[4] = {};
  // A fragment source: contiguous K chunks (0-15 low lanes, 16-31 high)
  const float* arowp = x + (size_t)(row0 + lrow) * E_ + hi * 16;

  for (int k0 = 0; k0 < E_; k0 += 32) {
    v16h a;
    const float* ap = arowp + k0;
#pragma unroll
    for (int i = 0; i < 8; ++i) {
      a[i]     = (f16)ap[i];
      a[i + 8] = (f16)ap[8 + i];
    }
#pragma unroll
    for (int t = 0; t < 4; ++t) {
      int m = nBase + t * 16 + lrow;
      const v16h* bp = (const v16h*)(Ztm + (size_t)m * E_ + k0 + hi * 16);
      acc[t] = __builtin_amdgcn_wmma_f32_16x16x32_f16(
          false, a, false, *bp, (short)0, acc[t], false, false);
    }
  }

  // epilogue: v = 2*zn*asinh(lam*xz/zn*cosh(2r) - (lam-1)*sinh(2r))
  float lamr[8];
#pragma unroll
  for (int j = 0; j < 8; ++j) lamr[j] = lamS[j + hi * 8];

  float vv[4][8];
  float rs[8];
#pragma unroll
  for (int j = 0; j < 8; ++j) rs[j] = 0.f;
#pragma unroll
  for (int t = 0; t < 4; ++t) {
    int m = nBase + t * 16 + lrow;
    float znv = znm[m], chv = chm[m], shv = shm[m];
    float izn = 1.f / znv;
#pragma unroll
    for (int j = 0; j < 8; ++j) {
      float lam = lamr[j];
      float g = lam * acc[t][j] * izn * chv - (lam - 1.f) * shv;
      float v = 2.f * znv * asinhf(g);
      vv[t][j] = v;
      rs[j] += v * v;
    }
  }
#pragma unroll
  for (int j = 0; j < 8; ++j) {
    float s = hred_add16(rs[j]);
    if (lrow == 0) atomicAdd(&v2S[j + hi * 8], s);
  }
  __syncthreads();

  float scl[8];
#pragma unroll
  for (int j = 0; j < 8; ++j)
    scl[j] = 1.f / (1.f + sqrtf(1.f + v2S[j + hi * 8]));

  int b = row0 >> 10;
  int s0g = row0 & (S_ - 1);

  float hsum[8];  // per-row, per-head (== this wave) sum of out^2
#pragma unroll
  for (int j = 0; j < 8; ++j) hsum[j] = 0.f;
#pragma unroll
  for (int t = 0; t < 4; ++t)
#pragma unroll
    for (int j = 0; j < 8; ++j) {
      float o = vv[t][j] * scl[j];
      vv[t][j] = o;
      hsum[j] += o * o;
    }
#pragma unroll
  for (int j = 0; j < 8; ++j) hsum[j] = hred_add16(hsum[j]);

  if (mat < 2) {
    float* P = (mat == 0) ? pq : pk;
    float* Q2 = (mat == 0) ? q2a : k2a;
    size_t basebh = ((size_t)b * H_ + wave) * S_;
#pragma unroll
    for (int t = 0; t < 4; ++t) {
      int hdl = t * 16 + lrow;
#pragma unroll
      for (int j = 0; j < 8; ++j) {
        int srow = s0g + j + hi * 8;
        P[(basebh + srow) * HD_ + hdl] = vv[t][j];
      }
    }
    if (lrow == 0) {
#pragma unroll
      for (int j = 0; j < 8; ++j) Q2[basebh + s0g + j + hi * 8] = hsum[j];
    }
  } else {
    size_t base80 = ((size_t)b * H_ + wave) * 80;
#pragma unroll
    for (int j = 0; j < 8; ++j) {
      float lamh = 2.f / fmaxf(1.f - hsum[j], EPSF);
      int srow = s0g + j + hi * 8;
#pragma unroll
      for (int t = 0; t < 4; ++t) {
        int hdl = t * 16 + lrow;
        Vt[(base80 + hdl) * S_ + srow] = (f16)(lamh * vv[t][j]);
      }
      if (lrow == 0) Vt[(base80 + 64) * S_ + srow] = (f16)(lamh - 1.f);
    }
  }
}

// ---------------------------------------------------------------------------
// Fused hyperbolic attention (flash-style, 1 wave per 16-query tile):
//   scores via WMMA QK^T -> arccosh distance -> online max/exp -> P transposed
//   through LDS -> WMMA against extended Vt (N=80; col 64 carries den).
//   Softmax 1/l cancels in num/den so no running-sum is kept.
// ---------------------------------------------------------------------------
__global__ __launch_bounds__(128) void hattn_kernel(
    const float* __restrict__ pq, const float* __restrict__ pk,
    const float* __restrict__ q2a, const float* __restrict__ k2a,
    const f16* __restrict__ Vt, float* __restrict__ out) {
  __shared__ f16 ptile[4][16][34];

  int tid = threadIdx.x, wave = tid >> 5, lane = tid & 31;
  int lrow = lane & 15, hi = lane >> 4;

  int g = blockIdx.x * 4 + wave;         // 2048 query tiles total
  int b = g >> 9;
  int h = (g >> 6) & 7;
  int qBase = (g & 63) * 16;

  size_t bh = (size_t)b * H_ + h;
  const float* pqb = pq + bh * S_ * HD_;
  const float* pkb = pk + bh * S_ * HD_;
  const float* q2b = q2a + bh * S_;
  const float* k2b = k2a + bh * S_;
  const f16* vtb = Vt + bh * 80 * S_;

  // Q A-fragments for K-dim = hd (two k-steps of 32)
  v16h aq0, aq1;
  {
    const float* ap = pqb + (size_t)(qBase + lrow) * HD_ + hi * 16;
#pragma unroll
    for (int i = 0; i < 8; ++i) {
      aq0[i]     = (f16)ap[i];
      aq0[i + 8] = (f16)ap[8 + i];
      aq1[i]     = (f16)ap[32 + i];
      aq1[i + 8] = (f16)ap[40 + i];
    }
  }
  float q2r[8], omq[8];
#pragma unroll
  for (int j = 0; j < 8; ++j) {
    q2r[j] = q2b[qBase + j + hi * 8];
    omq[j] = 1.f - q2r[j];
  }

  v8f acc[5] = {};
  float runM[8];
#pragma unroll
  for (int j = 0; j < 8; ++j) runM[j] = -3.0e38f;

  for (int kb = 0; kb < S_; kb += 32) {
    float sc[2][8];
#pragma unroll
    for (int hf = 0; hf < 2; ++hf) {
      int key = kb + hf * 16 + lrow;
      const float* kp = pkb + (size_t)key * HD_ + hi * 16;
      v16h bq0, bq1;
#pragma unroll
      for (int i = 0; i < 16; ++i) {
        bq0[i] = (f16)kp[i];
        bq1[i] = (f16)kp[32 + i];
      }
      v8f sacc = {};
      sacc = __builtin_amdgcn_wmma_f32_16x16x32_f16(false, aq0, false, bq0,
                                                    (short)0, sacc, false, false);
      sacc = __builtin_amdgcn_wmma_f32_16x16x32_f16(false, aq1, false, bq1,
                                                    (short)0, sacc, false, false);
      float k2v = k2b[key];
      float omk = 1.f - k2v;
#pragma unroll
      for (int j = 0; j < 8; ++j) {
        float d2 = fmaxf(q2r[j] + k2v - 2.f * sacc[j], 0.f);
        float dn = fmaxf(omq[j] * omk, EPSF);
        float arg = fmaxf(1.f + 2.f * d2 / dn, 1.f + EPSF);
        float dist = __logf(arg + sqrtf(arg * arg - 1.f));  // arccosh
        sc[hf][j] = -dist;
      }
    }
    // online rescale + exp, write P (f16) in C-layout into LDS
    float p0[8], p1[8];
#pragma unroll
    for (int j = 0; j < 8; ++j) {
      float tm = hred_max16(fmaxf(sc[0][j], sc[1][j]));
      float nm = fmaxf(runM[j], tm);
      float alpha = __expf(runM[j] - nm);
      runM[j] = nm;
      p0[j] = __expf(sc[0][j] - nm);
      p1[j] = __expf(sc[1][j] - nm);
#pragma unroll
      for (int t = 0; t < 5; ++t) acc[t][j] *= alpha;
    }
#pragma unroll
    for (int j = 0; j < 8; ++j) {
      int r = j + hi * 8;
      ptile[wave][r][lrow]      = (f16)p0[j];
      ptile[wave][r][lrow + 16] = (f16)p1[j];
    }
    __syncthreads();
    // re-read as A-fragment (transpose happens through LDS)
    v16h af;
    {
      int c = hi * 16;
#pragma unroll
      for (int i = 0; i < 8; ++i) {
        af[i]     = ptile[wave][lrow][c + i];
        af[i + 8] = ptile[wave][lrow][c + 8 + i];
      }
    }
    __syncthreads();
    // W x Vt_ext : N = 80, col 64 accumulates den = sum w*(lam-1)
#pragma unroll
    for (int t = 0; t < 5; ++t) {
      int hdg = t * 16 + lrow;
      const v16h* vp = (const v16h*)(vtb + (size_t)hdg * S_ + kb + hi * 16);
      acc[t] = __builtin_amdgcn_wmma_f32_16x16x32_f16(
          false, af, false, *vp, (short)0, acc[t], false, false);
    }
  }

  // epilogue: m_bar = num/den, half mobius scalar-mul, store [B,S,E]
#pragma unroll
  for (int j = 0; j < 8; ++j) {
    float den = __shfl(acc[4][j], hi * 16, 32);  // col 64 lives in lanes 0/16
    if (fabsf(den) < EPSF) den = EPSF;
    float iden = 1.f / den;
    float mb[4];
    float s2 = 0.f;
#pragma unroll
    for (int t = 0; t < 4; ++t) {
      mb[t] = acc[t][j] * iden;
      s2 += mb[t] * mb[t];
    }
    s2 = hred_add16(s2);
    float mn = fmaxf(sqrtf(s2), EPSF);
    float tt = fminf(mn, 1.f - 1e-6f);
    float at = 0.5f * __logf((1.f + tt) / (1.f - tt));  // arctanh
    float scale = tanhf(0.5f * at) / mn;
    int srow = qBase + j + hi * 8;
    float* orow = out + ((size_t)b * S_ + srow) * E_ + h * HD_;
#pragma unroll
    for (int t = 0; t < 4; ++t) orow[t * 16 + lrow] = scale * mb[t];
  }
}

// ---------------------------------------------------------------------------
extern "C" void kernel_launch(void* const* d_in, const int* in_sizes, int n_in,
                              void* d_out, int out_size, void* d_ws,
                              size_t ws_size, hipStream_t stream) {
  (void)in_sizes; (void)n_in; (void)out_size; (void)ws_size;
  const float* xq = (const float*)d_in[0];
  const float* xk = (const float*)d_in[1];
  const float* xv = (const float*)d_in[2];
  const float* zq = (const float*)d_in[3];
  const float* rq = (const float*)d_in[4];
  const float* zk = (const float*)d_in[5];
  const float* rk = (const float*)d_in[6];
  const float* zv = (const float*)d_in[7];
  const float* rv = (const float*)d_in[8];
  float* out = (float*)d_out;

  char* ws = (char*)d_ws;
  size_t off = 0;
  auto alloc = [&](size_t bytes) -> char* {
    char* p = ws + off;
    off += (bytes + 255) & ~(size_t)255;
    return p;
  };
  f16*   Zt = (f16*)alloc((size_t)3 * E_ * E_ * sizeof(f16));
  float* zn = (float*)alloc((size_t)3 * E_ * sizeof(float));
  float* ch = (float*)alloc((size_t)3 * E_ * sizeof(float));
  float* sh = (float*)alloc((size_t)3 * E_ * sizeof(float));
  float* pq = (float*)alloc((size_t)B_ * H_ * S_ * HD_ * sizeof(float));
  float* pk = (float*)alloc((size_t)B_ * H_ * S_ * HD_ * sizeof(float));
  float* q2 = (float*)alloc((size_t)B_ * H_ * S_ * sizeof(float));
  float* k2 = (float*)alloc((size_t)B_ * H_ * S_ * sizeof(float));
  f16*   Vt = (f16*)alloc((size_t)B_ * H_ * 80 * S_ * sizeof(f16));

  {
    size_t n = (size_t)3 * E_ * E_;
    prep_transpose_kernel<<<(unsigned)((n + 255) / 256), 256, 0, stream>>>(
        zq, zk, zv, Zt);
  }
  prep_params_kernel<<<(3 * E_ + 255) / 256, 256, 0, stream>>>(
      zq, zk, zv, rq, rk, rv, zn, ch, sh);
  {
    size_t n = (size_t)B_ * H_ * 80 * S_;
    prep_zero_kernel<<<(unsigned)((n + 255) / 256), 256, 0, stream>>>(Vt, n);
  }
  dim3 g1(256, 3);
  hlinear_kernel<<<g1, 256, 0, stream>>>(xq, xk, xv, Zt, zn, ch, sh, pq, pk,
                                         q2, k2, Vt);
  hattn_kernel<<<512, 128, 0, stream>>>(pq, pk, q2, k2, Vt, out);
}
